// GRU_26972394619445
// MI455X (gfx1250) — matrix operand order repaired and checked by reading, not compiled, not run on hardware
//
#include <hip/hip_runtime.h>
#include <hip/hip_bf16.h>
#include <math.h>

#define B_ 64
#define T_ 512
#define I_ 128
#define R_ 512
#define O_ 64

typedef __attribute__((ext_vector_type(16))) __bf16 v16bf;
typedef __attribute__((ext_vector_type(8)))  float  v8f;

union FragBF {
    v16bf v;
    uint4 q[2];
    unsigned short us[16];
};

__device__ __forceinline__ unsigned short f2bf(float f) {
    unsigned int u = __float_as_uint(f);
    u += 0x7FFFu + ((u >> 16) & 1u);   // round-to-nearest-even
    return (unsigned short)(u >> 16);
}

__device__ __forceinline__ v8f wmma_bf16(const FragBF& a, const FragBF& b, v8f c) {
    return __builtin_amdgcn_wmma_f32_16x16x32_bf16(
        false, a.v, false, b.v, (short)0, c, false, false);
}

// ---------------- f32 -> bf16 conversion ----------------
__global__ void cvt_f32_bf16(const float* __restrict__ src,
                             unsigned short* __restrict__ dst, int n) {
    int i = blockIdx.x * blockDim.x + threadIdx.x;
    if (i < n) dst[i] = f2bf(src[i]);
}

// Repack x[b][t][i] (f32) -> xbf[m][i] (bf16) with m = t*64 + b (time-major rows)
__global__ void pack_x_kernel(const float* __restrict__ x,
                              unsigned short* __restrict__ xbf) {
    int q = blockIdx.x * blockDim.x + threadIdx.x;
    if (q >= B_ * T_ * I_) return;
    int i = q & (I_ - 1);
    int m = q >> 7;          // t*64 + b
    int t = m >> 6, b = m & 63;
    xbf[q] = f2bf(x[(b * T_ + t) * I_ + i]);
}

// ---------------- input projections: xr, xz, xn ----------------
// M = 32768 (m = t*64+b), N = 512, K = 128.  One wave = 16x64 tile, one gate.
__global__ __launch_bounds__(256) void proj_kernel(
    const unsigned short* __restrict__ xbf,
    const unsigned short* __restrict__ Wr,   // [512][128] bf16
    const unsigned short* __restrict__ Wz,
    const unsigned short* __restrict__ Wn,
    const float* __restrict__ bn_x,
    float* __restrict__ xr_ws,               // [(t*64+b)][512]
    float* __restrict__ xz_ws,
    float* __restrict__ xn_out)              // hstore layout [(b*T+t)][512]
{
    const int lane = threadIdx.x & 31;
    const int lm = lane & 15, lh = lane >> 4;
    const int wg = blockIdx.x * 8 + (threadIdx.x >> 5);  // 0..49151
    const int gate = wg >> 14;                            // 3 * 16384
    const int rem = wg & 16383;
    const int m0 = (rem >> 3) * 16;
    const int n0 = (rem & 7) * 64;
    const unsigned short* W = (gate == 0) ? Wr : ((gate == 1) ? Wz : Wn);

    v8f acc[4] = {{}, {}, {}, {}};
#pragma unroll
    for (int kk = 0; kk < 4; ++kk) {
        const int k0 = kk * 32;
        FragBF a;
        const uint4* ap = (const uint4*)(xbf + (m0 + lm) * I_ + k0 + lh * 8);
        a.q[0] = ap[0];
        a.q[1] = ap[2];                     // +16 elements = +32B
#pragma unroll
        for (int j = 0; j < 4; ++j) {
            FragBF bm;
            const uint4* bp = (const uint4*)(W + (n0 + j * 16 + lm) * I_ + k0 + lh * 8);
            bm.q[0] = bp[0];
            bm.q[1] = bp[2];
            acc[j] = wmma_bf16(a, bm, acc[j]);
        }
    }
#pragma unroll
    for (int j = 0; j < 4; ++j)
#pragma unroll
        for (int v = 0; v < 8; ++v) {
            int m = m0 + v + 8 * lh;
            int n = n0 + j * 16 + lm;
            float val = acc[j][v];
            if (gate == 0)      xr_ws[m * R_ + n] = val;
            else if (gate == 1) xz_ws[m * R_ + n] = val;
            else {
                int b = m & 63, t = m >> 6;
                xn_out[(b * T_ + t) * R_ + n] = val + bn_x[n];
            }
        }
}

// ---------------- recurrent scan ----------------
// 4 blocks x 512 threads (16 waves).  Block g owns batch rows [16g, 16g+16).
// Wave w owns N-columns [32w, 32w+32) for ALL three gates -> gate fusion in regs.
// h kept f32 in registers (fragment layout) + bf16 double-buffered copy in LDS.
// K-loop software-pipelined with double-buffered A/B fragments; elementwise
// operands prefetched (global_prefetch_b8) at the top of the GEMM phase.
__global__ __launch_bounds__(512) void scan_kernel(
    const float* __restrict__ xr_ws, const float* __restrict__ xz_ws,
    const float* __restrict__ noise,
    const unsigned short* __restrict__ Wrh,  // [512][512] bf16
    const unsigned short* __restrict__ Wzh,
    const unsigned short* __restrict__ Wnh,
    const float* __restrict__ br, const float* __restrict__ bz,
    const float* __restrict__ bnh,
    float* __restrict__ hstore)              // xn on input at [b][t][*], h on output
{
    __shared__ __align__(16) unsigned short hbf[2][16 * 520];  // 520 = bank padding

    const int lane = threadIdx.x & 31;
    const int w = threadIdx.x >> 5;          // 0..15
    const int lm = lane & 15, lh = lane >> 4;
    const int b0 = blockIdx.x * 16;

    for (int i = threadIdx.x; i < 16 * 520; i += 512) hbf[0][i] = 0;

    float hreg[2][8];
#pragma unroll
    for (int j = 0; j < 2; ++j)
#pragma unroll
        for (int v = 0; v < 8; ++v) hreg[j][v] = 0.f;

    float brj[2], bzj[2], bnj[2];
#pragma unroll
    for (int j = 0; j < 2; ++j) {
        const int n = w * 32 + j * 16 + lm;
        brj[j] = br[n]; bzj[j] = bz[n]; bnj[j] = bnh[n];
    }

    // rolling base pointers; all per-(j,v) deltas become compile-time immediates
    const int nb = w * 32 + lm;              // n at j=0
    const int mb = 8 * lh;                   // m at v=0
    const size_t hBase = (size_t)(b0 + mb) * T_ * R_ + nb;   // + t*R_ per step
    const size_t xBase = (size_t)(b0 + mb) * R_ + nb;        // + t*(64*R_) per step

    // per-(gate, j) weight fragment base pointers (include lane's K sub-offset)
    const unsigned short* wb[6];
    {
        const unsigned short* Wg[3] = {Wrh, Wzh, Wnh};
#pragma unroll
        for (int g = 0; g < 3; ++g)
#pragma unroll
            for (int j = 0; j < 2; ++j)
                wb[g * 2 + j] = Wg[g] + (size_t)(w * 32 + j * 16 + lm) * R_ + lh * 8;
    }
    __syncthreads();

    for (int t = 0; t < T_; ++t) {
        const unsigned short* hb = hbf[t & 1];
        const float* pH  = hstore + hBase + (size_t)t * R_;        // xn in, h out
        const float* pN  = noise  + hBase + (size_t)t * R_;
        const float* pXr = xr_ws  + xBase + (size_t)t * (64 * R_);
        const float* pXz = xz_ws  + xBase + (size_t)t * (64 * R_);

        // prefetch this step's elementwise operands; ~whole GEMM phase of lead
#pragma unroll
        for (int v = 0; v < 8; ++v) {
            __builtin_prefetch(pXr + v * R_, 0, 3);
            __builtin_prefetch(pXz + v * R_, 0, 3);
            __builtin_prefetch(pH + (size_t)v * (T_ * R_), 0, 3);
            __builtin_prefetch(pN + (size_t)v * (T_ * R_), 0, 3);
        }

        v8f acc[3][2] = {};
        FragBF a[2], bm[2][6];

        // prologue: fragments for kk = 0
        {
            const uint4* ap = (const uint4*)(hb + lm * 520 + lh * 8);
            a[0].q[0] = ap[0];
            a[0].q[1] = ap[2];
#pragma unroll
            for (int u = 0; u < 6; ++u) {
                const uint4* bp = (const uint4*)(wb[u]);
                bm[0][u].q[0] = bp[0];
                bm[0][u].q[1] = bp[2];
            }
        }

#pragma unroll
        for (int kk = 0; kk < 16; ++kk) {
            const int cur = kk & 1, nxt = cur ^ 1;
            if (kk < 15) {                    // load kk+1 while kk computes
                const int k1 = (kk + 1) * 32;
                const uint4* ap = (const uint4*)(hb + lm * 520 + k1 + lh * 8);
                a[nxt].q[0] = ap[0];
                a[nxt].q[1] = ap[2];
#pragma unroll
                for (int u = 0; u < 6; ++u) {
                    const uint4* bp = (const uint4*)(wb[u] + k1);
                    bm[nxt][u].q[0] = bp[0];
                    bm[nxt][u].q[1] = bp[2];
                }
            }
#pragma unroll
            for (int u = 0; u < 6; ++u)
                acc[u >> 1][u & 1] = wmma_bf16(a[cur], bm[cur][u], acc[u >> 1][u & 1]);
        }

        // gate nonlinearities + h update, all in fragment layout / registers
        unsigned short* hw = hbf[(t + 1) & 1];
        float* pHo = (float*)pH;
#pragma unroll
        for (int j = 0; j < 2; ++j) {
#pragma unroll
            for (int v = 0; v < 8; ++v) {
                const float xr = pXr[v * R_ + j * 16];
                const float xz = pXz[v * R_ + j * 16];
                const float xn = pH[(size_t)v * (T_ * R_) + j * 16];  // staged by proj
                const float nz = pN[(size_t)v * (T_ * R_) + j * 16];
                const float r  = 1.f / (1.f + __expf(-(xr + acc[0][j][v] + brj[j])));
                const float z  = 1.f / (1.f + __expf(-(xz + acc[1][j][v] + bzj[j])));
                const float ng = tanhf(xn + r * (acc[2][j][v] + bnj[j]));
                const float h  = (1.f - z) * ng + z * hreg[j][v] + nz;
                hreg[j][v] = h;
                pHo[(size_t)v * (T_ * R_) + j * 16] = h;              // overwrite xn
                hw[(mb + v) * 520 + nb + j * 16] = f2bf(h);
            }
        }
        __syncthreads();   // double buffer -> single barrier per step
    }
}

// ---------------- output projection: y = hstore @ Wy^T + by ----------------
// M = 32768 (m = b*T+t, matches hstore rows), N = 64, K = 512.
__global__ __launch_bounds__(256) void yproj_kernel(
    const float* __restrict__ hstore,
    const unsigned short* __restrict__ Wybf,  // [64][512] bf16
    const float* __restrict__ by,
    float* __restrict__ y)
{
    const int lane = threadIdx.x & 31;
    const int lm = lane & 15, lh = lane >> 4;
    const int wgv = blockIdx.x * 8 + (threadIdx.x >> 5);  // 0..2047
    const int m0 = wgv * 16;

    float byv[4];
#pragma unroll
    for (int j = 0; j < 4; ++j) byv[j] = by[j * 16 + lm];

    v8f acc[4] = {{}, {}, {}, {}};
#pragma unroll 4
    for (int kk = 0; kk < 16; ++kk) {
        const int k0 = kk * 32;
        const float* arow = hstore + (m0 + lm) * R_ + k0 + lh * 8;
        const float4* af = (const float4*)arow;
        float4 f0 = af[0], f1 = af[1];      // 8 f32, K-chunk 1
        float4 f2 = af[4], f3 = af[5];      // 8 f32, K-chunk 2 (+16 cols)
        FragBF a;
        a.us[0] = f2bf(f0.x); a.us[1] = f2bf(f0.y); a.us[2] = f2bf(f0.z); a.us[3] = f2bf(f0.w);
        a.us[4] = f2bf(f1.x); a.us[5] = f2bf(f1.y); a.us[6] = f2bf(f1.z); a.us[7] = f2bf(f1.w);
        a.us[8]  = f2bf(f2.x); a.us[9]  = f2bf(f2.y); a.us[10] = f2bf(f2.z); a.us[11] = f2bf(f2.w);
        a.us[12] = f2bf(f3.x); a.us[13] = f2bf(f3.y); a.us[14] = f2bf(f3.z); a.us[15] = f2bf(f3.w);
#pragma unroll
        for (int j = 0; j < 4; ++j) {
            FragBF bm;
            const uint4* bp = (const uint4*)(Wybf + (j * 16 + lm) * R_ + k0 + lh * 8);
            bm.q[0] = bp[0];
            bm.q[1] = bp[2];
            acc[j] = wmma_bf16(a, bm, acc[j]);
        }
    }
#pragma unroll
    for (int j = 0; j < 4; ++j)
#pragma unroll
        for (int v = 0; v < 8; ++v) {
            const int m = m0 + v + 8 * lh;
            y[m * O_ + j * 16 + lm] = acc[j][v] + byv[j];
        }
}

// ---------------- host launcher ----------------
extern "C" void kernel_launch(void* const* d_in, const int* in_sizes, int n_in,
                              void* d_out, int out_size, void* d_ws, size_t ws_size,
                              hipStream_t stream) {
    const float* x     = (const float*)d_in[0];
    const float* noise = (const float*)d_in[1];
    const float* Wr_x  = (const float*)d_in[2];
    const float* Wr_h  = (const float*)d_in[3];
    const float* br    = (const float*)d_in[4];
    const float* Wz_x  = (const float*)d_in[5];
    const float* Wz_h  = (const float*)d_in[6];
    const float* bz    = (const float*)d_in[7];
    const float* Wn_x  = (const float*)d_in[8];
    const float* bn_x  = (const float*)d_in[9];
    const float* Wn_h  = (const float*)d_in[10];
    const float* bn_h  = (const float*)d_in[11];
    const float* Wy    = (const float*)d_in[12];
    const float* by    = (const float*)d_in[13];

    float* y_out = (float*)d_out;                       // [B,T,O]
    float* hstore = y_out + (size_t)B_ * T_ * O_;       // [B,T,R]

    // workspace carve-up
    float* xr_ws = (float*)d_ws;                        // 32768*512 f32
    float* xz_ws = xr_ws + (size_t)32768 * 512;
    unsigned short* x_bf = (unsigned short*)(xz_ws + (size_t)32768 * 512);
    unsigned short* Wrx = x_bf + (size_t)32768 * 128;
    unsigned short* Wzx = Wrx + 512 * 128;
    unsigned short* Wnx = Wzx + 512 * 128;
    unsigned short* Wrh = Wnx + 512 * 128;
    unsigned short* Wzh = Wrh + 512 * 512;
    unsigned short* Wnh = Wzh + 512 * 512;
    unsigned short* Wyb = Wnh + 512 * 512;

    // 1) weight conversions to bf16
    cvt_f32_bf16<<<(512 * 128 + 255) / 256, 256, 0, stream>>>(Wr_x, Wrx, 512 * 128);
    cvt_f32_bf16<<<(512 * 128 + 255) / 256, 256, 0, stream>>>(Wz_x, Wzx, 512 * 128);
    cvt_f32_bf16<<<(512 * 128 + 255) / 256, 256, 0, stream>>>(Wn_x, Wnx, 512 * 128);
    cvt_f32_bf16<<<(512 * 512 + 255) / 256, 256, 0, stream>>>(Wr_h, Wrh, 512 * 512);
    cvt_f32_bf16<<<(512 * 512 + 255) / 256, 256, 0, stream>>>(Wz_h, Wzh, 512 * 512);
    cvt_f32_bf16<<<(512 * 512 + 255) / 256, 256, 0, stream>>>(Wn_h, Wnh, 512 * 512);
    cvt_f32_bf16<<<(64 * 512 + 255) / 256, 256, 0, stream>>>(Wy, Wyb, 64 * 512);

    // 2) repack x (time-major, bf16)
    pack_x_kernel<<<(B_ * T_ * I_ + 255) / 256, 256, 0, stream>>>(x, x_bf);

    // 3) input projections (WMMA)
    proj_kernel<<<6144, 256, 0, stream>>>(x_bf, Wrx, Wzx, Wnx, bn_x,
                                          xr_ws, xz_ws, hstore);

    // 4) sequential GRU scan (WMMA, batch split over 4 WGPs, no grid sync)
    scan_kernel<<<4, 512, 0, stream>>>(xr_ws, xz_ws, noise, Wrh, Wzh, Wnh,
                                       br, bz, bn_h, hstore);

    // 5) output projection (WMMA)
    yproj_kernel<<<256, 256, 0, stream>>>(hstore, Wyb, by, y_out);

    (void)in_sizes; (void)n_in; (void)out_size; (void)ws_size;
}
